// ConcatenationAggregator_65575560675685
// MI455X (gfx1250) — matrix operand order repaired and checked by reading, not compiled, not run on hardware
//
#include <hip/hip_runtime.h>
#include <hip/hip_bf16.h>

typedef __attribute__((ext_vector_type(16))) _Float16 v16h;
typedef __attribute__((ext_vector_type(8)))  float    v8f;
typedef __attribute__((ext_vector_type(4)))  float    v4f;

#define N_REVIEWS 1000000
#define D_EMB     64
#define D_OUT     64
#define NCHUNK    6     // K = 192 = 6 * 32
#define NTILE     4     // N = 64  = 4 * 16
#define WAVES_PER_BLOCK 8

// Build one 16-bit A fragment (16x32, MxK) per wave.
// Per the CDNA5 ISA A-layout: lane group kh=lane>>4 holds two contiguous
// 8-float K-runs: [kh*8, kh*8+8) and [16+kh*8, 16+kh*8+8).
__device__ __forceinline__ v16h load_afrag(const float* __restrict__ src, int kh) {
    v4f a0 = *(const v4f*)(src + kh * 8);
    v4f a1 = *(const v4f*)(src + kh * 8 + 4);
    v4f a2 = *(const v4f*)(src + 16 + kh * 8);
    v4f a3 = *(const v4f*)(src + 16 + kh * 8 + 4);
    v16h A;
#pragma unroll
    for (int q = 0; q < 4; ++q) {
        A[q]      = (_Float16)a0[q];
        A[4 + q]  = (_Float16)a1[q];
        A[8 + q]  = (_Float16)a2[q];
        A[12 + q] = (_Float16)a3[q];
    }
    return A;
}

__global__ __launch_bounds__(256) void concat_agg_wmma_kernel(
    const float* __restrict__ review_vecs,   // [1e6, 64]
    const float* __restrict__ user_vecs,     // [1e5, 64]
    const float* __restrict__ item_vecs,     // [5e4, 64]
    const float* __restrict__ W,             // [192, 64]
    const int*   __restrict__ ua,            // [1e6]
    const int*   __restrict__ ia,            // [1e6]
    const int*   __restrict__ perm_u,        // [64]
    const int*   __restrict__ perm_i,        // [64]
    float*       __restrict__ out)           // [1e6, 64]
{
    // B fragments pre-swizzled into WMMA lane layout: [chunk][ntile][lane][16 halves]
    __shared__ __align__(32) _Float16 sB[NCHUNK * NTILE * 32 * 16];  // 24 KB
    __shared__ int sInv[128];  // invperm_u (0..63), invperm_i (64..127)

    const int tid = threadIdx.x;

    // Inverse permutations: fold the feature shuffle into W's row order.
    if (tid < 64)       sInv[perm_u[tid]] = tid;
    else if (tid < 128) sInv[64 + perm_i[tid - 64]] = tid - 64;
    __syncthreads();

    // Swizzle W (f32 -> f16) into B-fragment layout.
    // B (32x16, KxN): lane = kgroup*16 + n ; element j -> k = chunk*32 + (j>>3)*16 + kgroup*8 + (j&7).
    for (int e = tid; e < NCHUNK * NTILE * 32 * 16; e += 256) {
        int j    = e & 15;
        int lane = (e >> 4) & 31;
        int ct   = e >> 9;            // 0..23
        int t    = ct & 3;
        int c    = ct >> 2;
        int k    = c * 32 + ((j >> 3) << 4) + ((lane >> 4) << 3) + (j & 7);
        int n    = t * 16 + (lane & 15);
        int wrow;
        if (k < 64)       wrow = k;                           // review block: identity
        else if (k < 128) wrow = 64  + sInv[k - 64];          // user block: inv perm
        else              wrow = 128 + sInv[64 + (k - 128)];  // item block: inv perm
        sB[e] = (_Float16)W[wrow * D_OUT + n];
    }
    __syncthreads();

    const v16h* sBv = (const v16h*)sB;
    const int wave = tid >> 5;
    const int lane = tid & 31;
    const int m    = lane & 15;   // row-in-tile for A, column-in-ntile for C
    const int kh   = lane >> 4;   // lane half (K-group select)

    const int nTiles  = N_REVIEWS / 16;                 // 62500, exact
    const int wstride = gridDim.x * WAVES_PER_BLOCK;

    for (int tile = blockIdx.x * WAVES_PER_BLOCK + wave; tile < nTiles; tile += wstride) {
        const int row = tile * 16 + m;
        const int u   = ua[row];
        const int it  = ia[row];
        const float* s0 = review_vecs + (size_t)row * D_EMB;
        const float* s1 = user_vecs   + (size_t)u   * D_EMB;
        const float* s2 = item_vecs   + (size_t)it  * D_EMB;

        v8f acc0 = {}, acc1 = {}, acc2 = {}, acc3 = {};

#define KSTEP(c, srcptr)                                                              \
        {                                                                             \
            v16h A = load_afrag((srcptr), kh);                                        \
            v16h b0 = sBv[((c) * 4 + 0) * 32 + lane];                                 \
            acc0 = __builtin_amdgcn_wmma_f32_16x16x32_f16(false, A, false, b0,        \
                        (short)0, acc0, false, false);                                \
            v16h b1 = sBv[((c) * 4 + 1) * 32 + lane];                                 \
            acc1 = __builtin_amdgcn_wmma_f32_16x16x32_f16(false, A, false, b1,        \
                        (short)0, acc1, false, false);                                \
            v16h b2 = sBv[((c) * 4 + 2) * 32 + lane];                                 \
            acc2 = __builtin_amdgcn_wmma_f32_16x16x32_f16(false, A, false, b2,        \
                        (short)0, acc2, false, false);                                \
            v16h b3 = sBv[((c) * 4 + 3) * 32 + lane];                                 \
            acc3 = __builtin_amdgcn_wmma_f32_16x16x32_f16(false, A, false, b3,        \
                        (short)0, acc3, false, false);                                \
        }

        KSTEP(0, s0)        // review K 0..31
        KSTEP(1, s0 + 32)   // review K 32..63
        KSTEP(2, s1)        // user   K 0..31
        KSTEP(3, s1 + 32)   // user   K 32..63
        KSTEP(4, s2)        // item   K 0..31
        KSTEP(5, s2 + 32)   // item   K 32..63
#undef KSTEP

        // C/D layout: VGPR v -> M = v + 8*kh, N = (lane&15) within each 16-col ntile.
        const int obase = tile * 16 + kh * 8;
#pragma unroll
        for (int v = 0; v < 8; ++v) {
            float* op = out + (size_t)(obase + v) * D_OUT + m;
            op[0]  = fmaxf(acc0[v], 0.0f);
            op[16] = fmaxf(acc1[v], 0.0f);
            op[32] = fmaxf(acc2[v], 0.0f);
            op[48] = fmaxf(acc3[v], 0.0f);
        }
    }
}

extern "C" void kernel_launch(void* const* d_in, const int* in_sizes, int n_in,
                              void* d_out, int out_size, void* d_ws, size_t ws_size,
                              hipStream_t stream) {
    (void)in_sizes; (void)n_in; (void)out_size; (void)d_ws; (void)ws_size;
    const float* review = (const float*)d_in[0];
    const float* user   = (const float*)d_in[1];
    const float* item   = (const float*)d_in[2];
    const float* W      = (const float*)d_in[3];
    const int*   ua     = (const int*)d_in[4];
    const int*   ia     = (const int*)d_in[5];
    const int*   pu     = (const int*)d_in[6];
    const int*   pi     = (const int*)d_in[7];
    float* out = (float*)d_out;

    // 62500 tiles total; 8 waves/block. ~3125 blocks -> ~2-3 tiles per wave,
    // amortizing the per-block W swizzle while keeping plenty of parallelism.
    const int blocks = 3125;
    concat_agg_wmma_kernel<<<blocks, 256, 0, stream>>>(
        review, user, item, W, ua, ia, pu, pi, out);
}
